// MultiGAT_2413771621177
// MI455X (gfx1250) — matrix operand order repaired and checked by reading, not compiled, not run on hardware
//
#include <hip/hip_runtime.h>
#include <math.h>

typedef _Float16 v16h __attribute__((ext_vector_type(16)));
typedef _Float16 v8h  __attribute__((ext_vector_type(8)));
typedef float    v8f  __attribute__((ext_vector_type(8)));
typedef float    v4f  __attribute__((ext_vector_type(4)));

constexpr int BB    = 4;     // batch
constexpr int NN    = 2048;  // padded node count (2047 posts + 1 c-node)
constexpr int NREAL = 2047;
constexpr int DD    = 768;   // feature dim
constexpr int HH    = 6;     // heads
constexpr int DHH   = 128;   // head dim
constexpr float MASK_FILL = -1e-8f;

// Native CDNA5 transcendental tanh (V_TANH_F32) with fallbacks.
static __device__ inline float fast_tanh(float x) {
#if __has_builtin(__builtin_amdgcn_tanhf)
    return __builtin_amdgcn_tanhf(x);
#elif __has_builtin(__builtin_amdgcn_tanh_f32)
    return __builtin_amdgcn_tanh_f32(x);
#else
    float y;
    asm volatile("v_tanh_f32 %0, %1\n\ts_nop 1" : "=v"(y) : "v"(x));
    return y;
#endif
}

// ---------------------------------------------------------------------------
// concat(feature, c_node) -> cur (f32 residual copy) + curh (f16 GEMM operand)
__global__ __launch_bounds__(256) void k_build_feat(const float* __restrict__ feat,
                                                    const float* __restrict__ cnode,
                                                    float* __restrict__ cur,
                                                    _Float16* __restrict__ curh) {
    int idx = blockIdx.x * 256 + threadIdx.x;          // BB*NN*DD total
    int d = idx % DD;
    int n = (idx / DD) % NN;
    int b = idx / (DD * NN);
    float v = (n < NREAL) ? feat[((size_t)b * NREAL + n) * DD + d]
                          : cnode[(size_t)b * DD + d];
    cur[idx]  = v;
    curh[idx] = (_Float16)v;
}

// concat(p_mask, 1) -> pfull[b][n]
__global__ __launch_bounds__(256) void k_build_p(const float* __restrict__ pm,
                                                 float* __restrict__ pf) {
    int idx = blockIdx.x * 256 + threadIdx.x;          // BB*NN total
    int n = idx % NN;
    int b = idx / NN;
    pf[idx] = (n < NREAL) ? pm[(size_t)b * NREAL + n] : 1.0f;
}

// W (f32, 768x768) -> f16 copy, once per GAT module
__global__ __launch_bounds__(256) void k_cvt_w(const float* __restrict__ w,
                                               _Float16* __restrict__ wh) {
    int idx = blockIdx.x * 256 + threadIdx.x;          // DD*DD total
    wh[idx] = (_Float16)w[idx];
}

// ---------------------------------------------------------------------------
// qk = cur @ W^T   (per batch: 2048x768 @ 768x768), all-f16 operands.
// One wave computes a 32(m) x 64(e) C-tile: 2 A-frags x 4 B-frags = 8 WMMA/k-step.
// Emits f32 qk (for sq/sk) + transposed f16 qkt[b][h][dh][n] (attention B operand).
__global__ __launch_bounds__(256) void k_gemm_qk(const _Float16* __restrict__ curh,
                                                 const _Float16* __restrict__ Wh,
                                                 float* __restrict__ qk,
                                                 _Float16* __restrict__ qkt) {
    int wave = (blockIdx.x * 256 + threadIdx.x) >> 5;  // 4*64*12 = 3072 waves
    int lane = threadIdx.x & 31;
    int eg = wave % 12;                // 64-wide column group
    int mt = (wave / 12) % (NN / 32);  // 32-row tile
    int b  = wave / (12 * (NN / 32));
    int lm = lane & 15;
    int hi = lane >> 4;                // lane group 0/1

    v8f acc[2][4];
#pragma unroll
    for (int mi = 0; mi < 2; ++mi)
#pragma unroll
        for (int j = 0; j < 4; ++j) acc[mi][j] = (v8f)0.0f;

    const _Float16* arow0 = curh + ((size_t)b * NN + mt * 32 + lm) * DD;
    const _Float16* arow1 = arow0 + (size_t)16 * DD;

    for (int kt = 0; kt < DD / 32; ++kt) {
        int k0 = kt * 32;
        // ---- A fragments: lane = row m, halves are K contiguous (16*hi + 0..15) ----
        v8h a00 = *(const v8h*)(arow0 + k0 + 16 * hi);
        v8h a01 = *(const v8h*)(arow0 + k0 + 16 * hi + 8);
        v8h a10 = *(const v8h*)(arow1 + k0 + 16 * hi);
        v8h a11 = *(const v8h*)(arow1 + k0 + 16 * hi + 8);
        v16h af0, af1;
#pragma unroll
        for (int i = 0; i < 8; ++i) {
            af0[i] = a00[i]; af0[i + 8] = a01[i];
            af1[i] = a10[i]; af1[i + 8] = a11[i];
        }
        // ---- B fragments: lane = col n, halves are K contiguous ----
#pragma unroll
        for (int j = 0; j < 4; ++j) {
            int ecol = eg * 64 + j * 16 + lm;
            v16h bf = *(const v16h*)(Wh + (size_t)ecol * DD + k0 + 16 * hi);
            acc[0][j] = __builtin_amdgcn_wmma_f32_16x16x32_f16(
                false, af0, false, bf, (short)0, acc[0][j], false, false);
            acc[1][j] = __builtin_amdgcn_wmma_f32_16x16x32_f16(
                false, af1, false, bf, (short)0, acc[1][j], false, false);
        }
    }
    // ---- store: f32 row-major + f16 transposed per head ----
#pragma unroll
    for (int mi = 0; mi < 2; ++mi) {
#pragma unroll
        for (int j = 0; j < 4; ++j) {
            int ecol = eg * 64 + j * 16 + lm;
            int hh = ecol >> 7, dd = ecol & (DHH - 1);
#pragma unroll
            for (int i = 0; i < 8; ++i) {
                int row = mt * 32 + mi * 16 + i + 8 * hi;
                float v = acc[mi][j][i];
                qk[((size_t)b * NN + row) * DD + ecol] = v;
                qkt[(((size_t)b * HH + hh) * DHH + dd) * NN + row] = (_Float16)v;
            }
        }
    }
}

// ---------------------------------------------------------------------------
// sq[b,n,h] = qk[b,n,h,:].Wq[h]; sk likewise with Wk.
__global__ __launch_bounds__(256) void k_sqsk(const float* __restrict__ qk,
                                              const float* __restrict__ aw,
                                              float* __restrict__ sq,
                                              float* __restrict__ sk) {
    int t = blockIdx.x * 256 + threadIdx.x;            // BB*NN*HH total
    int h = t % HH;
    int n = (t / HH) % NN;
    int b = t / (HH * NN);
    const v4f* row = (const v4f*)(qk + ((size_t)b * NN + n) * DD + h * DHH);
    const v4f* wq  = (const v4f*)(aw + (size_t)h * 2 * DHH);
    const v4f* wk  = wq + DHH / 4;
    float s1 = 0.f, s2 = 0.f;
    for (int d = 0; d < DHH / 4; ++d) {
        v4f v = row[d], q = wq[d], k = wk[d];
#pragma unroll
        for (int c = 0; c < 4; ++c) { s1 += v[c] * q[c]; s2 += v[c] * k[c]; }
    }
    sq[t] = s1;
    sk[t] = s2;
}

// ---------------------------------------------------------------------------
// Fused attention: per (b,h,16-row q-tile) one wave streams k in 32-tiles,
// generates A = exp(masked tanh(sq+sk)) directly in WMMA A-layout, accumulates
// hid = A @ qk_h (d=128 via 8 WMMAs) and the softmax denominator, then writes
// feat += tanh(hid / denom) in place (residual fused, f32 + f16 copies).
// The per-tile sk/p staging uses the CDNA5 async cache->LDS path
// (global_load_async_to_lds_b32 + s_wait_asynccnt).
__global__ __launch_bounds__(32) void k_attn(float* __restrict__ cur,
                                             _Float16* __restrict__ curh,
                                             const _Float16* __restrict__ qkt,
                                             const float* __restrict__ sq,
                                             const float* __restrict__ sk,
                                             const float* __restrict__ pf) {
    int t  = blockIdx.x;                 // BB*HH*128 tasks
    int qt = t % (NN / 16);
    int h  = (t / (NN / 16)) % HH;
    int b  = t / ((NN / 16) * HH);
    int lane = threadIdx.x;
    int lm = lane & 15;
    int hi = lane >> 4;

    __shared__ float s_sk[32];
    __shared__ float s_p[32];
    __shared__ float s_den[16];

    int qrow = qt * 16 + lm;
    float sqv = sq[((size_t)b * NN + qrow) * HH + h];
    float pq  = pf[(size_t)b * NN + qrow];
    const _Float16* bbase = qkt + ((size_t)b * HH + h) * DHH * NN;

    // LDS byte offsets for this lane's async-load destinations
    unsigned ldst_sk = (unsigned)(size_t)&s_sk[lane];
    unsigned ldst_p  = (unsigned)(size_t)&s_p[lane];

    v8f acc[8];
#pragma unroll
    for (int j = 0; j < 8; ++j) acc[j] = (v8f)0.0f;
    float dsum = 0.f;

    for (int kt = 0; kt < NN / 32; ++kt) {
        int k0 = kt * 32;
        // ---- async global->LDS staging of this k-tile's sk / p values ----
        const float* gsk = sk + ((size_t)b * NN + k0 + lane) * HH + h;
        const float* gp  = pf + (size_t)b * NN + k0 + lane;
        asm volatile(
            "global_load_async_to_lds_b32 %0, %2, off\n\t"
            "global_load_async_to_lds_b32 %1, %3, off\n\t"
            "s_wait_asynccnt 0"
            :
            : "v"(ldst_sk), "v"(ldst_p), "v"(gsk), "v"(gp)
            : "memory");
        __syncthreads();

        v16h af;
#pragma unroll
        for (int i = 0; i < 16; ++i) {
            int kk = i + 16 * hi;   // 0..31, A layout
            float sc = (pq > 0.f && s_p[kk] > 0.f) ? fast_tanh(sqv + s_sk[kk])
                                                   : MASK_FILL;
            float e = __expf(sc);
            dsum += e;
            af[i] = (_Float16)e;
        }
#pragma unroll
        for (int j = 0; j < 8; ++j) {
            const _Float16* bp = bbase + (size_t)(j * 16 + lm) * NN + k0 + 16 * hi;
            v16h bf = *(const v16h*)bp;
            if (kt + 1 < NN / 32) __builtin_prefetch(bp + 32, 0, 1);
            acc[j] = __builtin_amdgcn_wmma_f32_16x16x32_f16(
                false, af, false, bf, (short)0, acc[j], false, false);
        }
        __syncthreads();
    }

    float den = dsum + __shfl_xor(dsum, 16, 32);  // merge lane l / l+16 row halves
    if (lane < 16) s_den[lane] = den;
    __syncthreads();

#pragma unroll
    for (int j = 0; j < 8; ++j) {
#pragma unroll
        for (int i = 0; i < 8; ++i) {
            int m = i + 8 * hi;
            int q = qt * 16 + m;
            int col = h * DHH + j * 16 + lm;
            size_t idx = ((size_t)b * NN + q) * DD + col;
            float val = cur[idx] + fast_tanh(acc[j][i] / s_den[m]);
            cur[idx]  = val;
            curh[idx] = (_Float16)val;
        }
    }
}

// ---------------------------------------------------------------------------
// MLP head (tiny): x=relu(out@W0^T+b0); x=relu(x@W1^T+b1); logits=x@fc^T+fb.
__global__ __launch_bounds__(256) void k_mlp(const float* __restrict__ cur,
                                             const float* __restrict__ w0,
                                             const float* __restrict__ bias0,
                                             const float* __restrict__ w1,
                                             const float* __restrict__ bias1,
                                             const float* __restrict__ fcw,
                                             const float* __restrict__ fcb,
                                             float* __restrict__ out) {
    __shared__ float sx[BB * DD];
    __shared__ float sh[BB * DD];
    int tid = threadIdx.x;
    for (int i = tid; i < BB * DD; i += 256) {
        int b = i / DD, d = i % DD;
        sx[i] = cur[((size_t)b * NN + NREAL) * DD + d];  // c-node row
    }
    __syncthreads();
    for (int i = tid; i < BB * DD; i += 256) {
        int b = i / DD, f = i % DD;
        float s = bias0[f];
        const float* w = w0 + (size_t)f * DD;
        const float* x = sx + b * DD;
        for (int d = 0; d < DD; ++d) s += x[d] * w[d];
        sh[i] = fmaxf(s, 0.f);
    }
    __syncthreads();
    for (int i = tid; i < BB * DD; i += 256) {
        int b = i / DD, f = i % DD;
        float s = bias1[f];
        const float* w = w1 + (size_t)f * DD;
        const float* x = sh + b * DD;
        for (int d = 0; d < DD; ++d) s += x[d] * w[d];
        sx[i] = fmaxf(s, 0.f);   // reuse sx for hidden-2
    }
    __syncthreads();
    if (tid < BB * 4) {
        int b = tid / 4, c = tid % 4;
        float s = fcb[c];
        const float* w = fcw + (size_t)c * DD;
        const float* x = sx + b * DD;
        for (int d = 0; d < DD; ++d) s += x[d] * w[d];
        out[b * 4 + c] = s;
    }
}

// ---------------------------------------------------------------------------
extern "C" void kernel_launch(void* const* d_in, const int* in_sizes, int n_in,
                              void* d_out, int out_size, void* d_ws, size_t ws_size,
                              hipStream_t stream) {
    const float* p_mask  = (const float*)d_in[0];
    const float* feature = (const float*)d_in[1];
    const float* c_node  = (const float*)d_in[2];
    const float* W_gat   = (const float*)d_in[3];
    const float* attn_w  = (const float*)d_in[4];
    const float* mlp_w0  = (const float*)d_in[5];
    const float* mlp_b0  = (const float*)d_in[6];
    const float* mlp_w1  = (const float*)d_in[7];
    const float* mlp_b1  = (const float*)d_in[8];
    const float* fc_w    = (const float*)d_in[9];
    const float* fc_b    = (const float*)d_in[10];
    float* out = (float*)d_out;

    char* ws = (char*)d_ws;
    size_t off = 0;
    auto carve = [&](size_t bytes) -> char* {
        char* p = ws + off;
        off += (bytes + 255) & ~(size_t)255;
        return p;
    };
    float*    cur   = (float*)   carve(sizeof(float)    * BB * NN * DD);       // 25.2 MB
    _Float16* curh  = (_Float16*)carve(sizeof(_Float16) * BB * NN * DD);       // 12.6 MB
    float*    qk    = (float*)   carve(sizeof(float)    * BB * NN * DD);       // 25.2 MB
    _Float16* qkt   = (_Float16*)carve(sizeof(_Float16) * BB * HH * DHH * NN); // 12.6 MB
    _Float16* Wh    = (_Float16*)carve(sizeof(_Float16) * DD * DD);            // 1.2 MB
    float*    sq    = (float*)   carve(sizeof(float)    * BB * NN * HH);
    float*    sk    = (float*)   carve(sizeof(float)    * BB * NN * HH);
    float*    pfull = (float*)   carve(sizeof(float)    * BB * NN);

    k_build_p<<<(BB * NN) / 256, 256, 0, stream>>>(p_mask, pfull);

    for (int g = 0; g < 4; ++g) {
        k_build_feat<<<(BB * NN * DD) / 256, 256, 0, stream>>>(feature, c_node, cur, curh);
        k_cvt_w<<<(DD * DD) / 256, 256, 0, stream>>>(W_gat + (size_t)g * DD * DD, Wh);
        for (int layer = 0; layer < 2; ++layer) {
            // 3072 waves / 8 per block = 384 blocks
            k_gemm_qk<<<384, 256, 0, stream>>>(curh, Wh, qk, qkt);
            k_sqsk<<<(BB * NN * HH) / 256, 256, 0, stream>>>(
                qk, attn_w + (size_t)g * HH * 2 * DHH, sq, sk);
            k_attn<<<BB * HH * (NN / 16), 32, 0, stream>>>(cur, curh, qkt, sq, sk, pfull);
        }
        k_mlp<<<1, 256, 0, stream>>>(cur, mlp_w0, mlp_b0, mlp_w1, mlp_b1,
                                     fc_w + (size_t)g * 4 * DD, fc_b + (size_t)g * 4,
                                     out + g * 16);
    }
}
